// ECN6_37391985279552
// MI455X (gfx1250) — matrix operand order repaired and checked by reading, not compiled, not run on hardware
//
#include <hip/hip_runtime.h>
#include <hip/hip_bf16.h>

typedef __bf16 bf16_t;
typedef __attribute__((ext_vector_type(16))) __bf16 v16bf;
typedef __attribute__((ext_vector_type(8)))  float   v8f;

#if defined(__has_builtin)
# if __has_builtin(__builtin_amdgcn_global_load_async_to_lds_b128)
#  define USE_ASYNC_LDS 1
# endif
#endif
#ifndef USE_ASYNC_LDS
# define USE_ASYNC_LDS 0
#endif

#if USE_ASYNC_LDS
// Exact pointee type from the builtin's signature: int vector_size(16),
// AS1 (printed as "__device__") for the global source, AS3 ("__shared__")
// for the LDS destination.
typedef int v4i_ __attribute__((vector_size(16)));
typedef __attribute__((address_space(1))) v4i_ as1_v4i;
typedef __attribute__((address_space(3))) v4i_ as3_v4i;

__device__ __forceinline__ void wait_asynccnt0() {
#if defined(__has_builtin)
# if __has_builtin(__builtin_amdgcn_s_wait_asynccnt)
    __builtin_amdgcn_s_wait_asynccnt(0);
    return;
# else
    asm volatile("s_wait_asynccnt 0x0" ::: "memory");
    return;
# endif
#else
    asm volatile("s_wait_asynccnt 0x0" ::: "memory");
#endif
}
#endif

// ---------------- problem constants (match setup_inputs) ----------------
#define BGR   256      // graphs
#define NPG   128      // nodes per graph
#define C0    53       // input features
#define KNN   4
#define N0    (BGR*NPG)      // 32768 nodes stage0
#define E1    (N0*KNN)       // 131072 edges
#define K1    103            // ceil(0.8*128)
#define N1V   (BGR*K1)       // 26368 nodes stage1
#define E2    (N1V*KNN)      // 105472
#define K2    83             // ceil(0.8*103)
#define N2V   (BGR*K2)       // 21248 nodes stage2
#define E3    (N2V*KNN)      // 84992
#define EPSBN 1e-5f

// ---------------- WMMA GEMM: out = epilogue(A[M,K]@Bt[N,K]^T) -----------
// mode 0: edge-conv epilogue: y=relu(z+bias)*scale+shift, mask per row,
//         mean over 4 consecutive rows -> out[node, col]
// mode 1: plain: out[row, col] = relu(z+bias)*scale+shift
#define MBLK 128
#define NBLK 64
#define KSTEP 32
#define LDS_STRIDE 40   // bf16 elems per LDS row (80B): conflict-free, 16B-aligned

__global__ __launch_bounds__(256) void wmma_gemm_kernel(
    const bf16_t* __restrict__ A, const bf16_t* __restrict__ Bt,
    float* __restrict__ out, int M, int N, int K,
    const float* __restrict__ bias, const float* __restrict__ scale,
    const float* __restrict__ shift, int mode,
    const float* __restrict__ emask, const float* __restrict__ invcnt)
{
    const int tid  = threadIdx.x;
    const int wave = tid >> 5;
    const int lane = tid & 31;
    const int mBase = blockIdx.y * MBLK;
    const int nBase = blockIdx.x * NBLK;

    v8f acc[4];
#pragma unroll
    for (int a = 0; a < 4; ++a)
#pragma unroll
        for (int i = 0; i < 8; ++i) acc[a][i] = 0.0f;

    const int arow   = tid >> 1;          // 0..127
    const int ahalf  = (tid & 1) << 4;    // 0 or 16 bf16
    const int brow   = tid >> 2;          // 0..63
    const int bchunk = (tid & 3) << 3;    // 0,8,16,24 bf16

#if USE_ASYNC_LDS
    // Double-buffered tiles filled by GLOBAL_LOAD_ASYNC_TO_LDS_B128 (ASYNCcnt).
    __shared__ bf16_t As[2][MBLK * LDS_STRIDE];
    __shared__ bf16_t Bs[2][NBLK * LDS_STRIDE];

    #define ISSUE_TILE(k0_, buf_)                                                      \
        do {                                                                           \
            const bf16_t* ag_ = A + (size_t)(mBase + arow) * K + (k0_) + ahalf;        \
            bf16_t* al_ = As[buf_] + arow * LDS_STRIDE + ahalf;                        \
            __builtin_amdgcn_global_load_async_to_lds_b128(                            \
                (as1_v4i*)ag_, (as3_v4i*)al_, 0, 0);                                   \
            __builtin_amdgcn_global_load_async_to_lds_b128(                            \
                (as1_v4i*)(ag_ + 8), (as3_v4i*)(al_ + 8), 0, 0);                       \
            const bf16_t* bg_ = Bt + (size_t)(nBase + brow) * K + (k0_) + bchunk;      \
            bf16_t* bl_ = Bs[buf_] + brow * LDS_STRIDE + bchunk;                       \
            __builtin_amdgcn_global_load_async_to_lds_b128(                            \
                (as1_v4i*)bg_, (as3_v4i*)bl_, 0, 0);                                   \
        } while (0)

    ISSUE_TILE(0, 0);
    int it = 0;
    for (int k0 = 0; k0 < K; k0 += KSTEP, ++it) {
        const int buf = it & 1;
        wait_asynccnt0();
        __syncthreads();                       // tile[buf] ready for every wave;
                                               // everyone done reading tile[buf^1]
        if (k0 + KSTEP < K) ISSUE_TILE(k0 + KSTEP, buf ^ 1);  // overlap with WMMA

        union { v16bf v; unsigned u[8]; } af;
        const unsigned* Ar = (const unsigned*)(As[buf] + (wave * 16 + (lane & 15)) * LDS_STRIDE);
        const int aoff = (lane >> 4) << 2;
#pragma unroll
        for (int v = 0; v < 4; ++v) { af.u[v] = Ar[aoff + v]; af.u[4 + v] = Ar[8 + aoff + v]; }

        const int boff = (lane >> 4) << 2;
#pragma unroll
        for (int a = 0; a < 4; ++a) {
            union { v16bf v; unsigned u[8]; } bfr;
            const unsigned* Br = (const unsigned*)(Bs[buf] + (a * 16 + (lane & 15)) * LDS_STRIDE);
#pragma unroll
            for (int v = 0; v < 4; ++v) { bfr.u[v] = Br[boff + v]; bfr.u[4 + v] = Br[8 + boff + v]; }
            acc[a] = __builtin_amdgcn_wmma_f32_16x16x32_bf16(
                false, af.v, false, bfr.v, (short)0, acc[a], false, false);
        }
    }
    #undef ISSUE_TILE
#else
    // Fallback: synchronous VGPR-staged tiles.
    __shared__ bf16_t As[MBLK * LDS_STRIDE];
    __shared__ bf16_t Bs[NBLK * LDS_STRIDE];
    for (int k0 = 0; k0 < K; k0 += KSTEP) {
        const uint4* ag = (const uint4*)(A + (size_t)(mBase + arow) * K + k0 + ahalf);
        uint4 a0 = ag[0];
        uint4 a1 = ag[1];
        const uint4* bg = (const uint4*)(Bt + (size_t)(nBase + brow) * K + k0 + bchunk);
        uint4 b0 = bg[0];
        *(uint4*)(As + arow * LDS_STRIDE + ahalf)      = a0;
        *(uint4*)(As + arow * LDS_STRIDE + ahalf + 8)  = a1;
        *(uint4*)(Bs + brow * LDS_STRIDE + bchunk)     = b0;
        __syncthreads();

        union { v16bf v; unsigned u[8]; } af;
        const unsigned* Ar = (const unsigned*)(As + (wave * 16 + (lane & 15)) * LDS_STRIDE);
        const int aoff = (lane >> 4) << 2;
#pragma unroll
        for (int v = 0; v < 4; ++v) { af.u[v] = Ar[aoff + v]; af.u[4 + v] = Ar[8 + aoff + v]; }

        const int boff = (lane >> 4) << 2;
#pragma unroll
        for (int a = 0; a < 4; ++a) {
            union { v16bf v; unsigned u[8]; } bfr;
            const unsigned* Br = (const unsigned*)(Bs + (a * 16 + (lane & 15)) * LDS_STRIDE);
#pragma unroll
            for (int v = 0; v < 4; ++v) { bfr.u[v] = Br[boff + v]; bfr.u[4 + v] = Br[8 + boff + v]; }
            acc[a] = __builtin_amdgcn_wmma_f32_16x16x32_bf16(
                false, af.v, false, bfr.v, (short)0, acc[a], false, false);
        }
        __syncthreads();
    }
#endif

    // D layout: element (m,n): lane = n + 16*(m>=8), vgpr = m&7
    const int rb = mBase + wave * 16 + ((lane >> 4) << 3);  // first of 8 rows in this lane
#pragma unroll
    for (int a = 0; a < 4; ++a) {
        const int col = nBase + a * 16 + (lane & 15);
        const float bi = bias ? bias[col] : 0.0f;
        const float sc = scale[col];
        const float sh = shift[col];
        if (mode == 0) {
            float s0 = 0.0f, s1 = 0.0f;
#pragma unroll
            for (int v = 0; v < 4; ++v) {
                float y0 = fmaxf(acc[a][v]     + bi, 0.0f) * sc + sh;
                float y1 = fmaxf(acc[a][4 + v] + bi, 0.0f) * sc + sh;
                s0 += y0 * (emask ? emask[rb + v]     : 1.0f);
                s1 += y1 * (emask ? emask[rb + 4 + v] : 1.0f);
            }
            const int node0 = rb >> 2;   // 4 edges per node, rb % 8 == 0
            out[(size_t)node0 * N + col]       = s0 * (invcnt ? invcnt[node0]     : 0.25f);
            out[(size_t)(node0 + 1) * N + col] = s1 * (invcnt ? invcnt[node0 + 1] : 0.25f);
        } else {
#pragma unroll
            for (int v = 0; v < 8; ++v)
                out[(size_t)(rb + v) * N + col] = fmaxf(acc[a][v] + bi, 0.0f) * sc + sh;
        }
    }
}

// ---------------- prep kernels ----------------
__global__ void prep_weight_kernel(const float* __restrict__ W, bf16_t* __restrict__ Wt,
                                   int K, int N, int Kp) {
    int t = blockIdx.x * blockDim.x + threadIdx.x;
    if (t >= N * Kp) return;
    int n = t / Kp, k = t - n * Kp;
    float v = (k < K) ? W[(size_t)k * N + n] : 0.0f;
    Wt[(size_t)n * Kp + k] = (bf16_t)v;
}

__global__ void prep_bn_kernel(const float* g, const float* be, const float* rm,
                               const float* rv, float* sc, float* sh, int C) {
    int c = blockIdx.x * blockDim.x + threadIdx.x;
    if (c >= C) return;
    float s = g[c] * rsqrtf(rv[c] + EPSBN);
    sc[c] = s;
    sh[c] = be[c] - rm[c] * s;
}

__global__ void prep_norm_kernel(const float* __restrict__ w, int C, float* __restrict__ o) {
    __shared__ float red[256];
    float s = 0.0f;
    for (int c = threadIdx.x; c < C; c += 256) s += w[c] * w[c];
    red[threadIdx.x] = s;
    __syncthreads();
    for (int st = 128; st > 0; st >>= 1) {
        if (threadIdx.x < st) red[threadIdx.x] += red[threadIdx.x + st];
        __syncthreads();
    }
    if (threadIdx.x == 0) o[0] = rsqrtf(red[0]);
}

// ---------------- kNN (k=4, per-graph, stable ties like lax.top_k) ------
__global__ void knn_kernel(const float* __restrict__ pos, int* __restrict__ src) {
    __shared__ float px[NPG], py[NPG], pz[NPG];
    const int b = blockIdx.x, i = threadIdx.x;
    const int gi = b * NPG + i;
    px[i] = pos[(size_t)gi * 3 + 0];
    py[i] = pos[(size_t)gi * 3 + 1];
    pz[i] = pos[(size_t)gi * 3 + 2];
    __syncthreads();
    float bd0 = 3e38f, bd1 = 3e38f, bd2 = 3e38f, bd3 = 3e38f;
    int   bj0 = 0, bj1 = 0, bj2 = 0, bj3 = 0;
    const float xi = px[i], yi = py[i], zi = pz[i];
    for (int j = 0; j < NPG; ++j) {
        if (j == i) continue;
        float dx = px[j] - xi, dy = py[j] - yi, dz = pz[j] - zi;
        float d = dx * dx + dy * dy + dz * dz;
        if (d < bd3) {
            if (d < bd2) { bd3 = bd2; bj3 = bj2;
                if (d < bd1) { bd2 = bd1; bj2 = bj1;
                    if (d < bd0) { bd1 = bd0; bj1 = bj0; bd0 = d; bj0 = j; }
                    else         { bd1 = d;   bj1 = j; } }
                else { bd2 = d; bj2 = j; } }
            else { bd3 = d; bj3 = j; }
        }
    }
    src[gi * 4 + 0] = b * NPG + bj0;
    src[gi * 4 + 1] = b * NPG + bj1;
    src[gi * 4 + 2] = b * NPG + bj2;
    src[gi * 4 + 3] = b * NPG + bj3;
}

// ---------------- TopK pool: stable rank == jax top_k order --------------
__global__ void pool_kernel(const float* __restrict__ h, int Cfeat,
                            const float* __restrict__ pw, const float* __restrict__ invn,
                            int n_cur, int k_keep,
                            int* __restrict__ perm, float* __restrict__ gate,
                            int* __restrict__ old2new) {
    __shared__ float scs[NPG];
    const int b = blockIdx.x, i = threadIdx.x;
    const int base = b * n_cur;
    if (i < n_cur) {
        float d = 0.0f;
        const float* row = h + (size_t)(base + i) * Cfeat;
        for (int c = 0; c < Cfeat; ++c) d += row[c] * pw[c];
        scs[i] = d * invn[0];
    }
    __syncthreads();
    if (i < n_cur) {
        float s = scs[i];
        int rank = 0;
        for (int j = 0; j < n_cur; ++j)
            rank += (scs[j] > s) || (scs[j] == s && j < i);
        if (rank < k_keep) {
            int v = b * k_keep + rank;
            perm[v] = base + i;
            gate[v] = tanhf(s);
            old2new[base + i] = v;
        } else {
            old2new[base + i] = -1;
        }
    }
}

// ---------------- edge-feature builders (bf16 GEMM A matrices) -----------
__global__ void build_edges1_kernel(const float* __restrict__ x, const int* __restrict__ src,
                                    bf16_t* __restrict__ A1) {
    const int e = blockIdx.x, c = threadIdx.x;   // 128 cols (Kp=128)
    const int i = e >> 2, j = src[e];
    float v = 0.0f;
    if (c < C0)            v = x[(size_t)i * C0 + c];
    else if (c < 2 * C0) { int cc = c - C0; v = x[(size_t)j * C0 + cc] - x[(size_t)i * C0 + cc]; }
    A1[(size_t)e * 128 + c] = (bf16_t)v;
}

__global__ void build_edges2_kernel(const float* __restrict__ h1, const int* __restrict__ src0,
                                    const int* __restrict__ perm1, const float* __restrict__ gate1,
                                    const int* __restrict__ old2new1,
                                    bf16_t* __restrict__ A2, float* __restrict__ mask2) {
    const int e = blockIdx.x, c = threadIdx.x;   // 256 cols
    const int v1 = e >> 2, kk = e & 3;
    const int i0 = perm1[v1];
    const int j0 = src0[i0 * 4 + kk];
    const int j1 = old2new1[j0];
    const int j1c = (j1 >= 0) ? j1 : 0;
    const float gi = gate1[v1], gj = gate1[j1c];
    const int jr = perm1[j1c];
    float v;
    if (c < 128) v = h1[(size_t)i0 * 128 + c] * gi;
    else {
        int cc = c - 128;
        v = h1[(size_t)jr * 128 + cc] * gj - h1[(size_t)i0 * 128 + cc] * gi;
    }
    A2[(size_t)e * 256 + c] = (bf16_t)v;
    if (c == 0) mask2[e] = (j1 >= 0) ? 1.0f : 0.0f;
}

__global__ void build_edges3_kernel(const float* __restrict__ h2, const int* __restrict__ src0,
                                    const int* __restrict__ perm1, const int* __restrict__ old2new1,
                                    const int* __restrict__ perm2, const float* __restrict__ gate2,
                                    const int* __restrict__ old2new2,
                                    bf16_t* __restrict__ A3, float* __restrict__ mask3) {
    const int e = blockIdx.x, c = threadIdx.x;   // 512 cols
    const int v2 = e >> 2, kk = e & 3;
    const int v1 = perm2[v2];
    const int v0 = perm1[v1];
    const int j0 = src0[v0 * 4 + kk];
    const int j1 = old2new1[j0];
    const int j2 = (j1 >= 0) ? old2new2[j1] : -1;
    const int j2c = (j2 >= 0) ? j2 : 0;
    const float gi = gate2[v2], gj = gate2[j2c];
    const int jr = perm2[j2c];
    float v;
    if (c < 256) v = h2[(size_t)v1 * 256 + c] * gi;
    else {
        int cc = c - 256;
        v = h2[(size_t)jr * 256 + cc] * gj - h2[(size_t)v1 * 256 + cc] * gi;
    }
    A3[(size_t)e * 512 + c] = (bf16_t)v;
    if (c == 0) mask3[e] = (j2 >= 0) ? 1.0f : 0.0f;
}

__global__ void invcnt_kernel(const float* __restrict__ mask, float* __restrict__ invcnt, int Nn) {
    int v = blockIdx.x * blockDim.x + threadIdx.x;
    if (v >= Nn) return;
    float c = mask[v * 4] + mask[v * 4 + 1] + mask[v * 4 + 2] + mask[v * 4 + 3];
    invcnt[v] = 1.0f / fmaxf(c, 1.0f);
}

// ---------------- global mean pool -> bf16 classifier input -------------
__global__ void gmp_kernel(const float* __restrict__ h3, bf16_t* __restrict__ gmp) {
    const int b = blockIdx.x, c = threadIdx.x;   // 512 threads
    float s = 0.0f;
    for (int r = 0; r < K2; ++r) s += h3[(size_t)(b * K2 + r) * 512 + c];
    gmp[(size_t)b * 512 + c] = (bf16_t)(s * (1.0f / (float)K2));
}

// ---------------- final 512->1 + BN + sigmoid ---------------------------
__global__ void final_kernel(const float* __restrict__ z1, const float* __restrict__ Wc2,
                             const float* __restrict__ bc2, const float* __restrict__ gc2,
                             const float* __restrict__ bec2, const float* __restrict__ rmc2,
                             const float* __restrict__ rvc2, float* __restrict__ out) {
    __shared__ float red[256];
    const int b = blockIdx.x, t = threadIdx.x;
    float s = 0.0f;
    for (int c = t; c < 512; c += 256) s += z1[(size_t)b * 512 + c] * Wc2[c];
    red[t] = s;
    __syncthreads();
    for (int st = 128; st > 0; st >>= 1) {
        if (t < st) red[t] += red[t + st];
        __syncthreads();
    }
    if (t == 0) {
        float z = fmaxf(red[0] + bc2[0], 0.0f);
        float sc = gc2[0] * rsqrtf(rvc2[0] + EPSBN);
        z = (z - rmc2[0]) * sc + bec2[0];
        out[b] = 1.0f / (1.0f + expf(-z));
    }
}

// ---------------- launch ----------------
extern "C" void kernel_launch(void* const* d_in, const int* in_sizes, int n_in,
                              void* d_out, int out_size, void* d_ws, size_t ws_size,
                              hipStream_t stream) {
    const float* x    = (const float*)d_in[0];
    const float* pos  = (const float*)d_in[1];
    const float* W1   = (const float*)d_in[2];
    const float* b1   = (const float*)d_in[3];
    const float* g1   = (const float*)d_in[4];
    const float* be1  = (const float*)d_in[5];
    const float* rm1  = (const float*)d_in[6];
    const float* rv1  = (const float*)d_in[7];
    const float* pw1  = (const float*)d_in[8];
    const float* W2   = (const float*)d_in[9];
    const float* b2   = (const float*)d_in[10];
    const float* g2   = (const float*)d_in[11];
    const float* be2  = (const float*)d_in[12];
    const float* rm2  = (const float*)d_in[13];
    const float* rv2  = (const float*)d_in[14];
    const float* pw2  = (const float*)d_in[15];
    const float* W3   = (const float*)d_in[16];
    const float* b3   = (const float*)d_in[17];
    const float* g3   = (const float*)d_in[18];
    const float* be3  = (const float*)d_in[19];
    const float* rm3  = (const float*)d_in[20];
    const float* rv3  = (const float*)d_in[21];
    const float* Wc1  = (const float*)d_in[22];
    const float* bc1  = (const float*)d_in[23];
    const float* gc1  = (const float*)d_in[24];
    const float* bec1 = (const float*)d_in[25];
    const float* rmc1 = (const float*)d_in[26];
    const float* rvc1 = (const float*)d_in[27];
    const float* Wc2  = (const float*)d_in[28];
    const float* bc2  = (const float*)d_in[29];
    const float* gc2  = (const float*)d_in[30];
    const float* bec2 = (const float*)d_in[31];
    const float* rmc2 = (const float*)d_in[32];
    const float* rvc2 = (const float*)d_in[33];
    float* out = (float*)d_out;

    // ---- workspace arena ----
    char* base = (char*)d_ws;
    size_t off = 0;
    auto alloc = [&](size_t bytes) -> void* {
        void* p = base + off;
        off = (off + bytes + 255) & ~((size_t)255);
        return p;
    };
    int*    src0     = (int*)   alloc((size_t)E1 * 4);
    int*    perm1    = (int*)   alloc((size_t)N1V * 4);
    float*  gate1    = (float*) alloc((size_t)N1V * 4);
    int*    o2n1     = (int*)   alloc((size_t)N0 * 4);
    int*    perm2    = (int*)   alloc((size_t)N2V * 4);
    float*  gate2    = (float*) alloc((size_t)N2V * 4);
    int*    o2n2     = (int*)   alloc((size_t)N1V * 4);
    float*  mask2    = (float*) alloc((size_t)E2 * 4);
    float*  mask3    = (float*) alloc((size_t)E3 * 4);
    float*  ic1      = (float*) alloc((size_t)N1V * 4);
    float*  ic2      = (float*) alloc((size_t)N2V * 4);
    bf16_t* W1t      = (bf16_t*)alloc((size_t)128 * 128 * 2);
    bf16_t* W2t      = (bf16_t*)alloc((size_t)256 * 256 * 2);
    bf16_t* W3t      = (bf16_t*)alloc((size_t)512 * 512 * 2);
    bf16_t* Wc1t     = (bf16_t*)alloc((size_t)512 * 512 * 2);
    float*  s1 = (float*)alloc(128 * 4); float* t1 = (float*)alloc(128 * 4);
    float*  s2 = (float*)alloc(256 * 4); float* t2 = (float*)alloc(256 * 4);
    float*  s3 = (float*)alloc(512 * 4); float* t3 = (float*)alloc(512 * 4);
    float*  sc1v = (float*)alloc(512 * 4); float* tc1v = (float*)alloc(512 * 4);
    float*  invn = (float*)alloc(2 * 4);
    bf16_t* gmp  = (bf16_t*)alloc((size_t)BGR * 512 * 2);
    float*  z1   = (float*) alloc((size_t)BGR * 512 * 4);
    // reused buffers
    float*  hbuf = (float*) alloc((size_t)N2V * 512 * 4);             // h1/h2/h3 (max 43.5MB)
    bf16_t* Abuf = (bf16_t*)alloc((size_t)E3 * 512 * 2);              // A1/A2/A3 (max 87MB)
    (void)ws_size; (void)in_sizes; (void)n_in; (void)out_size;

    // ---- prep: weights (transposed+padded bf16), BN fold, pool norms ----
    prep_weight_kernel<<<(128 * 128 + 255) / 256, 256, 0, stream>>>(W1, W1t, 106, 128, 128);
    prep_weight_kernel<<<(256 * 256 + 255) / 256, 256, 0, stream>>>(W2, W2t, 256, 256, 256);
    prep_weight_kernel<<<(512 * 512 + 255) / 256, 256, 0, stream>>>(W3, W3t, 512, 512, 512);
    prep_weight_kernel<<<(512 * 512 + 255) / 256, 256, 0, stream>>>(Wc1, Wc1t, 512, 512, 512);
    prep_bn_kernel<<<1, 128, 0, stream>>>(g1, be1, rm1, rv1, s1, t1, 128);
    prep_bn_kernel<<<1, 256, 0, stream>>>(g2, be2, rm2, rv2, s2, t2, 256);
    prep_bn_kernel<<<2, 256, 0, stream>>>(g3, be3, rm3, rv3, s3, t3, 512);
    prep_bn_kernel<<<2, 256, 0, stream>>>(gc1, bec1, rmc1, rvc1, sc1v, tc1v, 512);
    prep_norm_kernel<<<1, 256, 0, stream>>>(pw1, 128, invn + 0);
    prep_norm_kernel<<<1, 256, 0, stream>>>(pw2, 256, invn + 1);

    // ---- kNN graph ----
    knn_kernel<<<BGR, NPG, 0, stream>>>(pos, src0);

    // ---- EdgeConv1: [E1,128]bf16 @ W1t -> h1[N0,128] (fused masked mean) ----
    build_edges1_kernel<<<E1, 128, 0, stream>>>(x, src0, Abuf);
    wmma_gemm_kernel<<<dim3(128 / NBLK, E1 / MBLK), 256, 0, stream>>>(
        Abuf, W1t, hbuf, E1, 128, 128, b1, s1, t1, 0, nullptr, nullptr);

    // ---- Pool1 ----
    pool_kernel<<<BGR, NPG, 0, stream>>>(hbuf, 128, pw1, invn + 0, NPG, K1,
                                         perm1, gate1, o2n1);

    // ---- EdgeConv2 ----
    build_edges2_kernel<<<E2, 256, 0, stream>>>(hbuf, src0, perm1, gate1, o2n1, Abuf, mask2);
    invcnt_kernel<<<(N1V + 255) / 256, 256, 0, stream>>>(mask2, ic1, N1V);
    wmma_gemm_kernel<<<dim3(256 / NBLK, E2 / MBLK), 256, 0, stream>>>(
        Abuf, W2t, hbuf, E2, 256, 256, b2, s2, t2, 0, mask2, ic1);

    // ---- Pool2 ----
    pool_kernel<<<BGR, NPG, 0, stream>>>(hbuf, 256, pw2, invn + 1, K1, K2,
                                         perm2, gate2, o2n2);

    // ---- EdgeConv3 ----
    build_edges3_kernel<<<E3, 512, 0, stream>>>(hbuf, src0, perm1, o2n1,
                                                perm2, gate2, o2n2, Abuf, mask3);
    invcnt_kernel<<<(N2V + 255) / 256, 256, 0, stream>>>(mask3, ic2, N2V);
    wmma_gemm_kernel<<<dim3(512 / NBLK, E3 / MBLK), 256, 0, stream>>>(
        Abuf, W3t, hbuf, E3, 512, 512, b3, s3, t3, 0, mask3, ic2);

    // ---- global mean pool + classifier ----
    gmp_kernel<<<BGR, 512, 0, stream>>>(hbuf, gmp);
    wmma_gemm_kernel<<<dim3(512 / NBLK, BGR / MBLK), 256, 0, stream>>>(
        gmp, Wc1t, z1, BGR, 512, 512, bc1, sc1v, tc1v, 1, nullptr, nullptr);
    final_kernel<<<BGR, 256, 0, stream>>>(z1, Wc2, bc2, gc2, bec2, rmc2, rvc2, out);
}
